// MultiheadGQA_1194000908798
// MI455X (gfx1250) — matrix operand order repaired and checked by reading, not compiled, not run on hardware
//
#include <hip/hip_runtime.h>

typedef __bf16 bf16_t;
typedef __attribute__((ext_vector_type(8)))  __bf16 v8bf;
typedef __attribute__((ext_vector_type(16))) __bf16 v16bf;
typedef __attribute__((ext_vector_type(8)))  float  v8f;
typedef __attribute__((ext_vector_type(4)))  unsigned int u32x4;
typedef __attribute__((ext_vector_type(8)))  int i32x8;
typedef __attribute__((ext_vector_type(4)))  int i32x4;

#define WMMA_BF16(A, B, C) \
    __builtin_amdgcn_wmma_f32_16x16x32_bf16(false, (A), false, (B), (short)0, (C), false, false)

__device__ __forceinline__ v16bf frag_cat(v8bf lo, v8bf hi) {
    return __builtin_shufflevector(lo, hi, 0,1,2,3,4,5,6,7,8,9,10,11,12,13,14,15);
}

// A-matrix fragment (16x32 bf16), CDNA5 layout:
//   lane 0-15  : row M=lane,    K = {k0+0..7,  k0+16..23}
//   lane 16-31 : row M=lane-16, K = {k0+8..15, k0+24..31}
template<int STRIDE>
__device__ __forceinline__ v16bf load_frag_a(const bf16_t* base, int lane) {
    const int l16 = lane & 15, hi = (lane >> 4) & 1;
    const bf16_t* rp = base + l16 * STRIDE;
    v8bf a0 = *reinterpret_cast<const v8bf*>(rp + hi * 8);
    v8bf a1 = *reinterpret_cast<const v8bf*>(rp + 16 + hi * 8);
    return frag_cat(a0, a1);
}

// B-matrix fragment (32x16 bf16): column N striped across lanes, K split like A:
//   lane 0-15 : col N=lane, K={k0+0..7, k0+16..23} ; lane 16-31 : col N=lane-16, K={k0+8..15, k0+24..31}
// Tile stored [n][k] row-major with row stride STRIDE.
template<int STRIDE>
__device__ __forceinline__ v16bf load_frag_b(const bf16_t* base, int lane) {
    const int l16 = lane & 15, hi = (lane >> 4) & 1;
    const bf16_t* rp = base + l16 * STRIDE;
    v8bf b0 = *reinterpret_cast<const v8bf*>(rp + hi * 8);
    v8bf b1 = *reinterpret_cast<const v8bf*>(rp + 16 + hi * 8);
    return frag_cat(b0, b1);
}

// ---------------------------------------------------------------------------
// Tensor Data Mover: stage a [rows x 128] bf16 tile (row stride 128 elems in
// global) into LDS with +8 bf16 per-row padding (LDS row stride 136 = ASTR).
// D# per CDNA5 ISA:
//   group0: count=1 | lds_addr | global_addr[56:0] | type=2
//   group1: data_size=1(2B), pad_enable, pad_interval=5 (64 DW = one 256B row),
//           pad_amount=3 (4 DW = 8 bf16), tensor_dim0=128, tensor_dim1=2048,
//           tile_dim0=128, tile_dim1=rows, tensor_dim0_stride=128
// groups 2/3 zero (2-D tensor). Tracked with TENSORcnt.
// ---------------------------------------------------------------------------
__device__ __forceinline__ void tdm_load_tile(const bf16_t* gptr, void* lptr, int rows)
{
    const unsigned lds = (unsigned)(uintptr_t)lptr;  // low 32 bits = LDS offset
    const unsigned long long ga = (unsigned long long)(uintptr_t)gptr;
    u32x4 g0;
    g0[0] = 1u;                                     // count=1 (valid user D#)
    g0[1] = lds;                                    // lds_addr
    g0[2] = (unsigned)(ga & 0xffffffffu);           // global_addr[31:0]
    g0[3] = (unsigned)((ga >> 32) & 0x01ffffffu)    // global_addr[56:32]
          | (2u << 30);                             // type=2 ("image")
    i32x8 g1;
    g1[0] = (1 << 16)        // data_size = 1 -> 2 bytes
          | (1 << 20)        // pad_enable
          | (5 << 22)        // pad_interval: 64 DWORDs (= one 128-bf16 row)
          | (3 << 25);       // pad_amount: 4 DWORDs (= 8 bf16)
    g1[1] = (128 << 16);     // tensor_dim0[15:0] in [31:16]; atomic_addr=0
    g1[2] = (2048 << 16);    // tensor_dim0[31:16]=0 ; tensor_dim1[15:0]=2048
    g1[3] = (128 << 16);     // tensor_dim1[31:16]=0 ; tile_dim0=128
    g1[4] = rows & 0xffff;   // tile_dim1=rows ; tile_dim2=0
    g1[5] = 128;             // tensor_dim0_stride[31:0] = 128 elements
    g1[6] = 0;               // stride0[47:32]=0 ; stride1[15:0]=0
    g1[7] = 0;
    const i32x4 z4 = {0, 0, 0, 0};
#if defined(__clang_major__) && (__clang_major__ >= 23)
    const i32x8 z8 = {0, 0, 0, 0, 0, 0, 0, 0};
    __builtin_amdgcn_tensor_load_to_lds(g0, g1, z4, z4, z8, 0);
#else
    __builtin_amdgcn_tensor_load_to_lds(g0, g1, z4, z4, 0);
#endif
}

// ---------------------------------------------------------------------------
// GEMM: out[m,n] = (sum_k A[m,k] * W[n,k] + bias[n]) * scale
// A: [M,K] (float or bf16), W: [N,K] fp32 (torch Linear weight), M,N,K % 128 == 0
// PERMUTE: write bf16 out in [B,H,S,D] layout (B=4,H=16,S=2048,D=128)
// OUTF32 : write fp32 row-major (final projection)
// ---------------------------------------------------------------------------
#define GBM 128
#define GBN 128
#define GBK 32
#define GSTR (GBK + 8)

template<typename AT, bool PERMUTE, bool OUTF32>
__global__ __launch_bounds__(256)
void gemm_kernel(const AT* __restrict__ A, const float* __restrict__ W,
                 const float* __restrict__ bias,
                 bf16_t* __restrict__ out_bf, float* __restrict__ out_f32,
                 int M, int N, int K, float scale)
{
    __shared__ bf16_t As[GBM][GSTR];
    __shared__ bf16_t Bs[GBN][GSTR];

    const int tid  = threadIdx.x;
    const int lane = tid & 31;
    const int wid  = tid >> 5;        // 0..7
    const int wr   = wid & 3;         // 4 row groups of 32
    const int wc   = wid >> 2;        // 2 col groups of 64
    const int row0 = blockIdx.y * GBM;
    const int col0 = blockIdx.x * GBN;

    v8f acc[2][4];
    const v8f vzero = {0.f,0.f,0.f,0.f,0.f,0.f,0.f,0.f};
    #pragma unroll
    for (int i = 0; i < 2; ++i)
        #pragma unroll
        for (int j = 0; j < 4; ++j) acc[i][j] = vzero;

    const int lr = tid >> 1;            // 0..127
    const int lk = (tid & 1) * 16;      // 0 or 16

    for (int k0 = 0; k0 < K; k0 += GBK) {
        const AT*    srcA = A + (size_t)(row0 + lr) * K + k0 + lk;
        const float* srcW = W + (size_t)(col0 + lr) * K + k0 + lk;
        #pragma unroll
        for (int i = 0; i < 16; ++i) As[lr][lk + i] = (bf16_t)(float)srcA[i];
        #pragma unroll
        for (int i = 0; i < 16; ++i) Bs[lr][lk + i] = (bf16_t)srcW[i];
        __syncthreads();

        v16bf afrag[2], bfrag[4];
        #pragma unroll
        for (int i = 0; i < 2; ++i)
            afrag[i] = load_frag_a<GSTR>(&As[wr * 32 + i * 16][0], lane);
        #pragma unroll
        for (int j = 0; j < 4; ++j)
            bfrag[j] = load_frag_b<GSTR>(&Bs[wc * 64 + j * 16][0], lane);
        #pragma unroll
        for (int i = 0; i < 2; ++i)
            #pragma unroll
            for (int j = 0; j < 4; ++j)
                acc[i][j] = WMMA_BF16(afrag[i], bfrag[j], acc[i][j]);
        __syncthreads();
    }

    // Epilogue. C/D layout: VGPR v -> (lane<16 ? M=v,N=lane : M=v+8,N=lane-16)
    const int l16 = lane & 15, hi = (lane >> 4) & 1;
    #pragma unroll
    for (int i = 0; i < 2; ++i) {
        #pragma unroll
        for (int j = 0; j < 4; ++j) {
            const int mbase = row0 + wr * 32 + i * 16 + hi * 8;
            const int nc    = col0 + wc * 64 + j * 16 + l16;
            const float bv  = bias[nc];
            #pragma unroll
            for (int v = 0; v < 8; ++v) {
                const int m = mbase + v;
                const float val = (acc[i][j][v] + bv) * scale;
                if (OUTF32) {
                    out_f32[(size_t)m * N + nc] = val;
                } else if (PERMUTE) {
                    const int b = m >> 11, s = m & 2047;   // S = 2048
                    const int h = nc >> 7, d = nc & 127;   // D = 128
                    out_bf[(((size_t)(b * 16 + h) * 2048) + s) * 128 + d] = (bf16_t)val;
                } else {
                    out_bf[(size_t)m * N + nc] = (bf16_t)val;
                }
            }
        }
    }
}

// ---------------------------------------------------------------------------
// Flash attention: per (b,h) head, Q/K/V in [B*H, S, D] bf16, D=128, S=2048.
// Block = 128 threads (4 waves); each block: 64 q-rows; each wave: 16 q-rows.
// Q and (double-buffered) K tiles staged by the Tensor Data Mover.
// ---------------------------------------------------------------------------
#define AQ   64
#define AKV  64
#define AD   128
#define ASTR (AD + 8)    // 136 (must match TDM pad config)
#define VSTR (AKV + 8)   // 72

__global__ __launch_bounds__(128)
void attn_kernel(const bf16_t* __restrict__ Qp, const bf16_t* __restrict__ Kp,
                 const bf16_t* __restrict__ Vp, bf16_t* __restrict__ out)
{
    __shared__ bf16_t Qs[AQ][ASTR];
    __shared__ bf16_t Ks[2][AKV][ASTR];    // double-buffered via TDM prefetch
    __shared__ bf16_t Vt[AD][VSTR];        // V transposed: [d][kv]
    __shared__ bf16_t Ps[4][16][VSTR];     // per-wave P tile

    const int bh  = blockIdx.x;            // 0..63
    const int qt  = blockIdx.y;            // 0..31
    const int tid = threadIdx.x;
    const int lane = tid & 31, wid = tid >> 5;
    const int l16 = lane & 15, hi = (lane >> 4) & 1;
    const int NT = 2048 / AKV;

    const bf16_t* Qbase = Qp + ((size_t)bh * 2048 + qt * AQ) * AD;
    const bf16_t* Kbase = Kp + (size_t)bh * 2048 * AD;
    const bf16_t* Vbase = Vp + (size_t)bh * 2048 * AD;

    // TDM: stage Q tile and first K tile (wave 0 issues; TENSORcnt is per-wave)
    if (wid == 0) {
        tdm_load_tile(Qbase, &Qs[0][0], AQ);
        tdm_load_tile(Kbase, &Ks[0][0][0], AKV);
    }

    float m_i[8], l_i[8];
    v8f o[8];
    const v8f vzero = {0.f,0.f,0.f,0.f,0.f,0.f,0.f,0.f};
    #pragma unroll
    for (int i = 0; i < 8; ++i) { m_i[i] = -1e30f; l_i[i] = 0.f; o[i] = vzero; }

    for (int kt = 0; kt < NT; ++kt) {
        // stage V transposed (regular loads; TDM cannot transpose)
        for (int idx = tid; idx < AKV * AD; idx += 128) {
            const int r = idx >> 7, d = idx & 127;
            Vt[d][r] = Vbase[(size_t)(kt * AKV + r) * AD + d];
        }
        // prefetch next K tile; wait for the current one (in-order TDM: <=1
        // outstanding means everything before the prefetch has landed)
        if (wid == 0) {
            if (kt + 1 < NT) {
                tdm_load_tile(Kbase + (size_t)(kt + 1) * AKV * AD,
                              &Ks[(kt + 1) & 1][0][0], AKV);
                __builtin_amdgcn_s_wait_tensorcnt(1);
            } else {
                __builtin_amdgcn_s_wait_tensorcnt(0);
            }
        }
        __syncthreads();

        const bf16_t* Kcur = &Ks[kt & 1][0][0];

        // S = Q * K^T  (16 rows x 64 cols per wave), K-depth 128
        v8f s[4] = {vzero, vzero, vzero, vzero};
        #pragma unroll
        for (int kk = 0; kk < 4; ++kk) {
            const v16bf a = load_frag_a<ASTR>(&Qs[wid * 16][kk * 32], lane);
            #pragma unroll
            for (int ct = 0; ct < 4; ++ct) {
                const v16bf b = load_frag_b<ASTR>(Kcur + (ct * 16) * ASTR + kk * 32, lane);
                s[ct] = WMMA_BF16(a, b, s[ct]);
            }
        }

        // online softmax (rows live inside one 16-lane half -> shfl_xor reduce)
        float mn[8], alpha[8];
        #pragma unroll
        for (int i = 0; i < 8; ++i) {
            float t = fmaxf(fmaxf(s[0][i], s[1][i]), fmaxf(s[2][i], s[3][i]));
            t = fmaxf(t, __shfl_xor(t, 1, 32));
            t = fmaxf(t, __shfl_xor(t, 2, 32));
            t = fmaxf(t, __shfl_xor(t, 4, 32));
            t = fmaxf(t, __shfl_xor(t, 8, 32));
            mn[i]    = fmaxf(m_i[i], t);
            alpha[i] = expf(m_i[i] - mn[i]);
        }
        #pragma unroll
        for (int ct = 0; ct < 4; ++ct)
            #pragma unroll
            for (int i = 0; i < 8; ++i)
                s[ct][i] = expf(s[ct][i] - mn[i]);
        #pragma unroll
        for (int i = 0; i < 8; ++i) {
            float t = s[0][i] + s[1][i] + s[2][i] + s[3][i];
            t += __shfl_xor(t, 1, 32);
            t += __shfl_xor(t, 2, 32);
            t += __shfl_xor(t, 4, 32);
            t += __shfl_xor(t, 8, 32);
            l_i[i] = l_i[i] * alpha[i] + t;
            m_i[i] = mn[i];
        }
        #pragma unroll
        for (int j = 0; j < 8; ++j)
            #pragma unroll
            for (int i = 0; i < 8; ++i) o[j][i] *= alpha[i];

        // P: C-layout -> LDS (re-layout for A-fragment reads)
        #pragma unroll
        for (int ct = 0; ct < 4; ++ct)
            #pragma unroll
            for (int i = 0; i < 8; ++i)
                Ps[wid][i + hi * 8][ct * 16 + l16] = (bf16_t)s[ct][i];
        __syncthreads();

        // O += P * V  (K-depth 64 over kv positions)
        #pragma unroll
        for (int kk = 0; kk < 2; ++kk) {
            const v16bf a = load_frag_a<VSTR>(&Ps[wid][0][kk * 32], lane);
            #pragma unroll
            for (int ct = 0; ct < 8; ++ct) {
                const v16bf b = load_frag_b<VSTR>(&Vt[ct * 16][kk * 32], lane);
                o[ct] = WMMA_BF16(a, b, o[ct]);
            }
        }
        __syncthreads();   // protect Vt (and retired K buffer) before next tile
    }

    // epilogue: normalize by row sum, write [B,S,E] bf16
    const int b = bh >> 4, h = bh & 15;
    #pragma unroll
    for (int ct = 0; ct < 8; ++ct) {
        #pragma unroll
        for (int i = 0; i < 8; ++i) {
            const int r = i + hi * 8;
            const int q = qt * AQ + wid * 16 + r;
            const int c = h * 128 + ct * 16 + l16;
            out[((size_t)(b * 2048 + q)) * 2048 + c] = (bf16_t)(o[ct][i] / l_i[i]);
        }
    }
}

// ---------------------------------------------------------------------------
// LayerNorm over E=2048, one block per row, 256 threads x 8 elems.
// ---------------------------------------------------------------------------
__global__ __launch_bounds__(256)
void ln_kernel(const bf16_t* __restrict__ x, const float* __restrict__ g,
               const float* __restrict__ b, bf16_t* __restrict__ y)
{
    __shared__ float red[2][8];
    const int row = blockIdx.x;
    const int tid = threadIdx.x;
    const bf16_t* xr = x + (size_t)row * 2048;

    float v[8], s = 0.f, ss = 0.f;
    #pragma unroll
    for (int i = 0; i < 8; ++i) {
        v[i] = (float)xr[tid + i * 256];
        s += v[i]; ss += v[i] * v[i];
    }
    #pragma unroll
    for (int m = 1; m < 32; m <<= 1) {
        s  += __shfl_xor(s,  m, 32);
        ss += __shfl_xor(ss, m, 32);
    }
    const int wid = tid >> 5, lane = tid & 31;
    if (lane == 0) { red[0][wid] = s; red[1][wid] = ss; }
    __syncthreads();
    float ts = 0.f, tss = 0.f;
    #pragma unroll
    for (int i = 0; i < 8; ++i) { ts += red[0][i]; tss += red[1][i]; }
    const float mu   = ts  * (1.f / 2048.f);
    const float var  = tss * (1.f / 2048.f) - mu * mu;
    const float rstd = rsqrtf(var + 1e-5f);
    #pragma unroll
    for (int i = 0; i < 8; ++i) {
        const int c = tid + i * 256;
        y[(size_t)row * 2048 + c] = (bf16_t)((v[i] - mu) * rstd * g[c] + b[c]);
    }
}

// ---------------------------------------------------------------------------
extern "C" void kernel_launch(void* const* d_in, const int* in_sizes, int n_in,
                              void* d_out, int out_size, void* d_ws, size_t ws_size,
                              hipStream_t stream)
{
    const float* query = (const float*)d_in[0];
    const float* key   = (const float*)d_in[1];
    const float* value = (const float*)d_in[2];
    const float* Wq = (const float*)d_in[3];
    const float* bq = (const float*)d_in[4];
    const float* Wk = (const float*)d_in[5];
    const float* bk = (const float*)d_in[6];
    const float* Wv = (const float*)d_in[7];
    const float* bv = (const float*)d_in[8];
    const float* ln_g = (const float*)d_in[9];
    const float* ln_b = (const float*)d_in[10];
    const float* Wo = (const float*)d_in[11];
    const float* bo = (const float*)d_in[12];
    float* out = (float*)d_out;

    const int M = 4 * 2048, N = 2048, K = 2048;
    const size_t NE = (size_t)M * N;   // 16,777,216 elements

    bf16_t* Qp   = (bf16_t*)d_ws;      // [B,H,S,D] bf16
    bf16_t* Kp   = Qp + NE;
    bf16_t* Vp   = Kp + NE;
    bf16_t* attn = Vp + NE;            // [B,S,E] bf16
    bf16_t* lnb  = attn + NE;          // [B,S,E] bf16

    dim3 ggrid(N / GBN, M / GBM);      // (16, 64)
    const float qscale = 0.08838834764831845f;  // 1/sqrt(128)

    gemm_kernel<float, true,  false><<<ggrid, 256, 0, stream>>>(
        query, Wq, bq, Qp, nullptr, M, N, K, qscale);
    gemm_kernel<float, true,  false><<<ggrid, 256, 0, stream>>>(
        key,   Wk, bk, Kp, nullptr, M, N, K, 1.0f);
    gemm_kernel<float, true,  false><<<ggrid, 256, 0, stream>>>(
        value, Wv, bv, Vp, nullptr, M, N, K, 1.0f);

    attn_kernel<<<dim3(64, 32), 128, 0, stream>>>(Qp, Kp, Vp, attn);

    ln_kernel<<<M, 256, 0, stream>>>(attn, ln_g, ln_b, lnb);

    gemm_kernel<bf16_t, false, true><<<ggrid, 256, 0, stream>>>(
        lnb, Wo, bo, nullptr, out, M, N, K, 1.0f);
}
